// SelfAttention_9405978378346
// MI455X (gfx1250) — matrix operand order repaired and checked by reading, not compiled, not run on hardware
//
#include <hip/hip_runtime.h>

#define DM   1024      // d_model
#define SEQ  1024      // sequence length
#define NB   8         // batch
#define NH   16        // heads
#define DK   64        // d_k
#define MROWS (NB * SEQ)   // 8192 rows for all projections

typedef __attribute__((ext_vector_type(16))) __bf16 v16bf;
typedef __attribute__((ext_vector_type(8)))  float  v8f;

union Frag { unsigned u[8]; v16bf v; };

// float -> bf16 (round-to-nearest-even), stored as ushort
__device__ __forceinline__ unsigned short f2bf(float f) {
  union { float f; unsigned u; } c; c.f = f;
  unsigned r = c.u + 0x7FFFu + ((c.u >> 16) & 1u);
  return (unsigned short)(r >> 16);
}

// A-matrix (16x32 bf16) element->K mapping (contiguous, matches B fills):
// VGPR j: K=2j,2j+1 (+16*half)
__device__ __forceinline__ int a_kbase(int j, int half) {
  return 2 * j + 16 * half;
}

__device__ __forceinline__ v8f wmma_bf16(const Frag& a, const Frag& b, v8f c) {
  return __builtin_amdgcn_wmma_f32_16x16x32_bf16(false, a.v, false, b.v,
                                                 (short)0, c, false, false);
}

// ---- async global->LDS copy (CDNA5 GLOBAL_LOAD_ASYNC_TO_LDS_B32) ----------
#if __has_builtin(__builtin_amdgcn_global_load_async_to_lds_b32)
#define HAVE_ASYNC_LDS 1
typedef __attribute__((address_space(1))) int as1_int;
typedef __attribute__((address_space(3))) int as3_int;
__device__ __forceinline__ void async_cp_b32(const void* g, void* l) {
  __builtin_amdgcn_global_load_async_to_lds_b32((as1_int*)g, (as3_int*)l, 0, 0);
}
__device__ __forceinline__ void async_wait0() {
  __asm__ volatile("s_wait_asynccnt 0" ::: "memory");
}
#else
#define HAVE_ASYNC_LDS 0
#endif

// ---------------------------------------------------------------------------
// GEMM: C[M,N] = A[M,K] @ Bw[K,N] + bias, A is f32 or bf16(ushort),
// Bw is f32 (converted to bf16 while staging). Output optionally f32 and/or bf16.
// Block tile 128x128, 8 waves (4 in M x 2 in N), wave tile 32x64
// (2 A-frags x 4 B-frags -> 8 WMMA per 32-deep K-step).
// ---------------------------------------------------------------------------
#define BM 128
#define BN 128
#define BK 32
#define LDS_S (BK + 8)   // padded LDS row stride (bf16 elems)

template <bool A_IS_F32>
__global__ void __launch_bounds__(256)
gemm_wmma_kernel(const void* __restrict__ Aptr, const float* __restrict__ Bw,
                 const float* __restrict__ bias, float* __restrict__ Cf,
                 unsigned short* __restrict__ Cb, int M, int N, int K) {
  __shared__ unsigned short As[BM][LDS_S];   // [m][k]
  __shared__ unsigned short Bs[BN][LDS_S];   // [n][k]  (stored transposed)

  const int t = threadIdx.x;
  const int wave = t >> 5, lane = t & 31;
  const int lm = lane & 15, half = lane >> 4;
  const int m0 = blockIdx.y * BM, n0 = blockIdx.x * BN;
  const int wm = (wave & 3) * 32;    // 4 waves across M
  const int wn = (wave >> 2) * 64;   // 2 waves across N

  const float* Af = (const float*)Aptr;
  const unsigned short* Ab = (const unsigned short*)Aptr;

  v8f c[2][4];
  for (int i = 0; i < 2; ++i)
    for (int j = 0; j < 4; ++j)
      for (int r = 0; r < 8; ++r) c[i][j][r] = 0.f;

  for (int kt = 0; kt < K; kt += BK) {
    // stage A tile 128x32
    if (A_IS_F32) {
      for (int q = 0; q < (BM * BK) / 256; ++q) {   // 16
        int i = t + 256 * q;
        int r = i >> 5, cc = i & 31;
        As[r][cc] = f2bf(Af[(size_t)(m0 + r) * K + kt + cc]);
      }
    } else {
#if HAVE_ASYNC_LDS
      for (int q = 0; q < (BM * BK) / 512; ++q) {   // 8 b32 copies/thread
        int d = t + 256 * q;                        // dword id 0..2047
        int r = d >> 4, c2 = (d & 15) * 2;
        async_cp_b32(Ab + (size_t)(m0 + r) * K + kt + c2, &As[r][c2]);
      }
#else
      for (int q = 0; q < (BM * BK) / 256; ++q) {
        int i = t + 256 * q;
        int r = i >> 5, cc = i & 31;
        As[r][cc] = Ab[(size_t)(m0 + r) * K + kt + cc];
      }
#endif
    }
    // stage B tile 32x128 transposed into [n][k], converting f32->bf16
    for (int q = 0; q < (BN * BK) / 256; ++q) {     // 16
      int i = t + 256 * q;
      int kk = i >> 7, nn = i & 127;
      Bs[nn][kk] = f2bf(Bw[(size_t)(kt + kk) * N + n0 + nn]);
    }
#if HAVE_ASYNC_LDS
    if (!A_IS_F32) async_wait0();
#endif
    __syncthreads();

    Frag a[2], b[4];
    for (int i = 0; i < 2; ++i)
      for (int j = 0; j < 8; ++j)
        a[i].u[j] = *(const unsigned*)&As[wm + 16 * i + lm][a_kbase(j, half)];
    for (int i = 0; i < 4; ++i)
      for (int j = 0; j < 8; ++j)
        b[i].u[j] = *(const unsigned*)&Bs[wn + 16 * i + lm][16 * half + 2 * j];
    for (int i = 0; i < 2; ++i)
      for (int j = 0; j < 4; ++j)
        c[i][j] = wmma_bf16(a[i], b[j], c[i][j]);
    __syncthreads();
  }

  // epilogue: C layout lane=col (N=lm), rows r + 8*half
  for (int i = 0; i < 2; ++i)
    for (int j = 0; j < 4; ++j) {
      int col = n0 + wn + 16 * j + lm;
      float bv = bias ? bias[col] : 0.f;
      for (int r = 0; r < 8; ++r) {
        int row = m0 + wm + 16 * i + r + 8 * half;
        float val = c[i][j][r] + bv;
        if (Cf) Cf[(size_t)row * N + col] = val;
        if (Cb) Cb[(size_t)row * N + col] = f2bf(val);
      }
    }
}

// ---------------------------------------------------------------------------
// Flash attention: grid (SEQ/128, NB*NH). 8 waves/block, each wave owns 16
// query rows. Keys processed in chunks of 32 staged in LDS by the block.
// ---------------------------------------------------------------------------
#define KC 32

__global__ void __launch_bounds__(256)
attn_wmma_kernel(const unsigned short* __restrict__ Qb,
                 const unsigned short* __restrict__ Kb,
                 const unsigned short* __restrict__ Vb,
                 const int* __restrict__ mask,
                 unsigned short* __restrict__ ctx) {
  __shared__ unsigned short Ks[KC][DK + 8];       // [key][dk]
  __shared__ unsigned short Vt[DK][KC + 8];       // [dk][key]  (transposed)
  __shared__ unsigned short Ps[8][16][KC + 8];    // per-wave P transpose buffer

  const int t = threadIdx.x, wave = t >> 5, lane = t & 31;
  const int lm = lane & 15, half = lane >> 4;
  const int bb = blockIdx.y >> 4, hh = blockIdx.y & 15;
  const int q0 = blockIdx.x * 128 + wave * 16;

  // load Q fragments for this wave's 16 rows (dk 0..31 and 32..63)
  Frag qf[2];
  {
    const unsigned short* qrow =
        Qb + ((size_t)(bb * SEQ + q0 + lm) * DM + hh * DK);
    for (int ks = 0; ks < 2; ++ks)
      for (int j = 0; j < 8; ++j)
        qf[ks].u[j] = *(const unsigned*)(qrow + ks * 32 + a_kbase(j, half));
  }

  float mi[8], li[8];
  v8f acc[4];
  for (int r = 0; r < 8; ++r) { mi[r] = -1e30f; li[r] = 0.f; }
  for (int j = 0; j < 4; ++j)
    for (int r = 0; r < 8; ++r) acc[j][r] = 0.f;

  const float scale = 0.03125f;  // 1/sqrt(D_MODEL) = 1/32

  for (int kc = 0; kc < SEQ; kc += KC) {
    __syncthreads();  // previous chunk's LDS reads done
    // stage K chunk [32][64]: pure bf16 copy -> async DMA path if available
#if HAVE_ASYNC_LDS
    for (int q = 0; q < (KC * DK) / 512; ++q) {   // 4 b32 copies/thread
      int d = t + 256 * q;                        // dword id 0..1023
      int key = d >> 5, dk = (d & 31) * 2;
      async_cp_b32(Kb + (size_t)(bb * SEQ + kc + key) * DM + hh * DK + dk,
                   &Ks[key][dk]);
    }
#else
    for (int q = 0; q < (KC * DK) / 256; ++q) {
      int i = t + 256 * q;
      int key = i >> 6, dk = i & 63;
      Ks[key][dk] = Kb[(size_t)(bb * SEQ + kc + key) * DM + hh * DK + dk];
    }
#endif
    // stage transposed V chunk [64][32] (layout change -> VGPR bounce)
    for (int q = 0; q < (KC * DK) / 256; ++q) {   // 8
      int i = t + 256 * q;
      int key = i >> 6, dk = i & 63;
      Vt[dk][key] = Vb[(size_t)(bb * SEQ + kc + key) * DM + hh * DK + dk];
    }
#if HAVE_ASYNC_LDS
    async_wait0();
#endif
    __syncthreads();

    // scores S = Q @ K^T  (two 16x16 key sub-tiles, K-dim = dk = 64)
    v8f s[2];
    for (int ns = 0; ns < 2; ++ns) {
      Frag k0, k1;
      const unsigned short* krow = &Ks[ns * 16 + lm][0];
      for (int j = 0; j < 8; ++j) {
        k0.u[j] = *(const unsigned*)&krow[16 * half + 2 * j];        // dk 0..31
        k1.u[j] = *(const unsigned*)&krow[32 + 16 * half + 2 * j];   // dk 32..63
      }
      v8f z;
      for (int r = 0; r < 8; ++r) z[r] = 0.f;
      z = wmma_bf16(qf[0], k0, z);
      z = wmma_bf16(qf[1], k1, z);
      s[ns] = z;
    }

    // scale + mask (mask is per (b, key))
    for (int ns = 0; ns < 2; ++ns) {
      int mv = mask[bb * SEQ + kc + ns * 16 + lm];
      for (int r = 0; r < 8; ++r) {
        float x = s[ns][r] * scale;
        s[ns][r] = (mv == 0) ? -3.125e18f : x;
      }
    }

    // online softmax: row reductions across the 16 lanes of each half
    float mx[8];
    for (int r = 0; r < 8; ++r) mx[r] = fmaxf(s[0][r], s[1][r]);
    for (int d = 1; d < 16; d <<= 1)
      for (int r = 0; r < 8; ++r)
        mx[r] = fmaxf(mx[r], __shfl_xor(mx[r], d, 32));

    float alpha[8], rs[8];
    for (int r = 0; r < 8; ++r) {
      float mnew = fmaxf(mi[r], mx[r]);
      alpha[r] = __expf(mi[r] - mnew);
      float p0 = __expf(s[0][r] - mnew);
      float p1 = __expf(s[1][r] - mnew);
      s[0][r] = p0; s[1][r] = p1;
      rs[r] = p0 + p1;
      mi[r] = mnew;
    }
    for (int d = 1; d < 16; d <<= 1)
      for (int r = 0; r < 8; ++r) rs[r] += __shfl_xor(rs[r], d, 32);
    for (int r = 0; r < 8; ++r) li[r] = li[r] * alpha[r] + rs[r];
    for (int j = 0; j < 4; ++j)
      for (int r = 0; r < 8; ++r) acc[j][r] *= alpha[r];

    // transpose P (C-layout lane=key -> A-layout lane=row) through wave-private LDS
    for (int ns = 0; ns < 2; ++ns)
      for (int r = 0; r < 8; ++r)
        Ps[wave][r + 8 * half][ns * 16 + lm] = f2bf(s[ns][r]);
    __asm__ volatile("s_wait_dscnt 0" ::: "memory");  // per-wave DS RAW fence

    Frag pf;
    for (int j = 0; j < 8; ++j)
      pf.u[j] = *(const unsigned*)&Ps[wave][lm][a_kbase(j, half)];

    // acc += P (16x32) @ V (32x64) over 4 dk sub-tiles
    for (int j = 0; j < 4; ++j) {
      Frag vf;
      const unsigned short* vrow = &Vt[16 * j + lm][0];
      for (int jj = 0; jj < 8; ++jj)
        vf.u[jj] = *(const unsigned*)&vrow[16 * half + 2 * jj];
      acc[j] = wmma_bf16(pf, vf, acc[j]);
    }
  }

  // normalize and store context in [b, q, h, dk] bf16 layout
  for (int j = 0; j < 4; ++j) {
    int dk = 16 * j + lm;
    for (int r = 0; r < 8; ++r) {
      int q = q0 + r + 8 * half;
      float o = acc[j][r] / li[r];
      ctx[(size_t)(bb * SEQ + q) * DM + hh * DK + dk] = f2bf(o);
    }
  }
}

// ---------------------------------------------------------------------------
extern "C" void kernel_launch(void* const* d_in, const int* in_sizes, int n_in,
                              void* d_out, int out_size, void* d_ws,
                              size_t ws_size, hipStream_t stream) {
  (void)in_sizes; (void)n_in; (void)out_size; (void)ws_size;
  const float* values = (const float*)d_in[0];
  const float* keys   = (const float*)d_in[1];
  const float* query  = (const float*)d_in[2];
  const int*   mask   = (const int*)d_in[3];
  const float* Wq = (const float*)d_in[4];
  const float* bq = (const float*)d_in[5];
  const float* Wk = (const float*)d_in[6];
  const float* bk = (const float*)d_in[7];
  const float* Wv = (const float*)d_in[8];
  const float* bv = (const float*)d_in[9];
  const float* Wo = (const float*)d_in[10];
  const float* bo = (const float*)d_in[11];
  float* out = (float*)d_out;

  // workspace: Q, K, V, context as bf16 (16 MB each, 64 MB total)
  const size_t elems = (size_t)MROWS * DM;
  unsigned short* Qb = (unsigned short*)d_ws;
  unsigned short* Kb = Qb + elems;
  unsigned short* Vb = Kb + elems;
  unsigned short* Cx = Vb + elems;

  dim3 blk(256);
  dim3 gg(DM / BN, MROWS / BM);  // (8, 64)
  gemm_wmma_kernel<true><<<gg, blk, 0, stream>>>(query,  Wq, bq, nullptr, Qb,
                                                 MROWS, DM, DM);
  gemm_wmma_kernel<true><<<gg, blk, 0, stream>>>(keys,   Wk, bk, nullptr, Kb,
                                                 MROWS, DM, DM);
  gemm_wmma_kernel<true><<<gg, blk, 0, stream>>>(values, Wv, bv, nullptr, Vb,
                                                 MROWS, DM, DM);

  dim3 ga(SEQ / 128, NB * NH);   // (8, 128)
  attn_wmma_kernel<<<ga, blk, 0, stream>>>(Qb, Kb, Vb, mask, Cx);

  gemm_wmma_kernel<false><<<gg, blk, 0, stream>>>(Cx, Wo, bo, out, nullptr,
                                                  MROWS, DM, DM);
}
